// SoftALU_62380105007441
// MI455X (gfx1250) — matrix operand order repaired and checked by reading, not compiled, not run on hardware
//
#include <hip/hip_runtime.h>

// CDNA5 / gfx1250 wave32 WMMA types
typedef __attribute__((ext_vector_type(16))) _Float16 v16h;
typedef __attribute__((ext_vector_type(8)))  float    v8f;
typedef __attribute__((ext_vector_type(4)))  float    f32x4;

#define BATCH 8192
#define OPSTRIDE ((size_t)BATCH * 4 * 256)   // one op-plane of the (7,8192,4,256) output

// Decode one 256-wide one-hot row to its byte value using v_wmma_f32_16x16x32_f16.
// `row` already includes the per-lane K-half offset (lanes>=16 read K=16..31 of each
// 32-chunk, in this file's contiguous K order: column N = lane%16, K = (lane<16?0:16)+i).
// `w0` is the A-matrix weight vector for chunk 0: A[m][k] = k (identical rows), laid
// out in the same contiguous K order as B. Each chunk adds 32 to the weights.
// After 8 chunks, D[m][n] = decoded value of element n for every m, so acc[0]
// holds element (lane%16)'s value in every lane. Exact: 0/1 and <=255 are exact f16.
__device__ __forceinline__ unsigned decode_byte(const float* __restrict__ row, v16h w0) {
  v16h w = w0;
  v8f acc = {};
  #pragma unroll
  for (int t = 0; t < 8; ++t) {
    const f32x4* p = (const f32x4*)(row + t * 32);
    f32x4 f0 = __builtin_nontemporal_load(p + 0);
    f32x4 f1 = __builtin_nontemporal_load(p + 1);
    f32x4 f2 = __builtin_nontemporal_load(p + 2);
    f32x4 f3 = __builtin_nontemporal_load(p + 3);
    v16h b;
    b[0]  = (_Float16)f0.x; b[1]  = (_Float16)f0.y; b[2]  = (_Float16)f0.z; b[3]  = (_Float16)f0.w;
    b[4]  = (_Float16)f1.x; b[5]  = (_Float16)f1.y; b[6]  = (_Float16)f1.z; b[7]  = (_Float16)f1.w;
    b[8]  = (_Float16)f2.x; b[9]  = (_Float16)f2.y; b[10] = (_Float16)f2.z; b[11] = (_Float16)f2.w;
    b[12] = (_Float16)f3.x; b[13] = (_Float16)f3.y; b[14] = (_Float16)f3.z; b[15] = (_Float16)f3.w;
    // 8 args: (neg_a, A, neg_b, B, c_mod, C, reuse_a, reuse_b)
    acc = __builtin_amdgcn_wmma_f32_16x16x32_f16(false, w, false, b, (short)0, acc,
                                                 false, false);
    #pragma unroll
    for (int i = 0; i < 16; ++i) w[i] = w[i] + (_Float16)32.0f;  // next chunk offset
  }
  return (unsigned)(acc[0] + 0.5f);
}

__global__ void __launch_bounds__(256)
softalu_kernel(const float* __restrict__ A, const float* __restrict__ Bm,
               float* __restrict__ O) {
  const int lane = threadIdx.x & 31;
  const int wid  = (int)((blockIdx.x * blockDim.x + threadIdx.x) >> 5);  // 0..511
  const int e    = (wid << 4) + (lane & 15);   // batch element owned by this lane
  const int half = lane >> 4;                  // which 128-col half this lane owns

  // A-matrix weights for chunk 0: contiguous K order matching B, element i -> K=b+i,
  // with b = 16 for lanes 16..31. All rows identical => value
  // independent of M, so c[0] is valid for every lane.
  v16h w0;
  {
    const int b = half ? 16 : 0;
    #pragma unroll
    for (int i = 0; i < 16; ++i) {
      const int k = b + i;
      w0[i] = (_Float16)(float)k;
    }
  }

  const int kb = half ? 16 : 0;  // B-operand K-half offset within each 32-chunk
  unsigned abyte[4], bbyte[4];
  #pragma unroll
  for (int n = 0; n < 4; ++n) {
    const size_t roff = ((size_t)e * 4 + n) * 256 + kb;
    abyte[n] = decode_byte(A  + roff, w0);
    bbyte[n] = decode_byte(Bm + roff, w0);
  }
  const unsigned av = abyte[0] | (abyte[1] << 8) | (abyte[2] << 16) | (abyte[3] << 24);
  const unsigned bv = bbyte[0] | (bbyte[1] << 8) | (bbyte[2] << 16) | (bbyte[3] << 24);

  // The seven exact integer results (soft pipeline saturates to these one-hots).
  unsigned r[7];
  r[0] = av + bv;                       // soft_add
  r[1] = av - bv;                       // soft_add(a, soft_negate(b)) == a - b mod 2^32
  r[2] = av * bv;                       // soft_mul (u32 wrap)
  r[3] = (bv == 0u) ? 0u : (av / bv);   // soft_div with guarded zero
  r[4] = av & bv;                       // AND table
  r[5] = av | bv;                       // OR table
  r[6] = av ^ bv;                       // XOR table

  // One-hot re-encode: each lane zero-fills its 128-float half with NT b128 stores,
  // then drops a single 1.0f at the hot column (same-lane stores stay ordered).
  const f32x4 z = {0.0f, 0.0f, 0.0f, 0.0f};
  #pragma unroll
  for (int op = 0; op < 7; ++op) {
    #pragma unroll
    for (int n = 0; n < 4; ++n) {
      float* row = O + (size_t)op * OPSTRIDE + (((size_t)e * 4 + (size_t)n) << 8)
                     + ((size_t)half << 7);
      #pragma unroll 4
      for (int j = 0; j < 32; ++j)
        __builtin_nontemporal_store(z, ((f32x4*)row) + j);
      const unsigned p = (r[op] >> (8 * n)) & 255u;
      if ((int)(p >> 7) == half)
        row[p & 127u] = 1.0f;
    }
  }
}

extern "C" void kernel_launch(void* const* d_in, const int* in_sizes, int n_in,
                              void* d_out, int out_size, void* d_ws, size_t ws_size,
                              hipStream_t stream) {
  (void)in_sizes; (void)n_in; (void)out_size; (void)d_ws; (void)ws_size;
  const float* a = (const float*)d_in[0];
  const float* b = (const float*)d_in[1];
  float* out = (float*)d_out;
  // 8192 elements / 16 per wave = 512 waves = 64 blocks x 8 waves (256 threads, wave32)
  hipLaunchKernelGGL(softalu_kernel, dim3(64), dim3(256), 0, stream, a, b, out);
}
